// CrossModalCodebook_10204842295879
// MI455X (gfx1250) — matrix operand order repaired and checked by reading, not compiled, not run on hardware
//
#include <hip/hip_runtime.h>
#include <hip/hip_bf16.h>

// ---------- types ----------
typedef __attribute__((ext_vector_type(16))) __bf16          v16bf;
typedef __attribute__((ext_vector_type(8)))  float           v8f;
typedef __attribute__((ext_vector_type(8)))  unsigned short  v8us;
typedef __attribute__((ext_vector_type(16))) unsigned short  v16us;
typedef __attribute__((ext_vector_type(4)))  unsigned int    v4ui;
typedef __attribute__((ext_vector_type(8)))  int             v8i;
typedef __attribute__((ext_vector_type(4)))  int             v4i;

#define DEV __device__ __forceinline__

// fp32 -> bf16, round-to-nearest-even
DEV unsigned short f2bf(float f) {
    unsigned int u = __float_as_uint(f);
    unsigned int r = (u + 0x7FFFu + ((u >> 16) & 1u)) >> 16;
    return (unsigned short)r;
}

// monotone float ordering in unsigned bits (packed argmin keys)
DEV unsigned int fbits_ordered(float f) {
    unsigned int b = __float_as_uint(f);
    return (b & 0x80000000u) ? ~b : (b | 0x80000000u);
}

// A-fragment (16x32 bf16): lane m=lane&15;
// elems 0..15 = K klo..klo+15, klo = 16*(lane>>4). p -> row base + klo.
DEV v16bf load_frag_a(const unsigned short* p) {
    v8us lo = *(const v8us*)(p);
    v8us hi = *(const v8us*)(p + 8);
    v16us u;
#pragma unroll
    for (int i = 0; i < 8; ++i) { u[i] = lo[i]; u[i + 8] = hi[i]; }
    return __builtin_bit_cast(v16bf, u);
}

// B-fragment (32x16 bf16, stored N-major): lane n=lane&15;
// elems 0..15 = K klo..klo+15, klo = 16*(lane>>4). p -> row base + klo.
DEV v16bf load_frag_b(const unsigned short* p) {
    v8us lo = *(const v8us*)(p);
    v8us hi = *(const v8us*)(p + 8);
    v16us u;
#pragma unroll
    for (int i = 0; i < 8; ++i) { u[i] = lo[i]; u[i + 8] = hi[i]; }
    return __builtin_bit_cast(v16bf, u);
}

DEV v8f wmma_bf16(v16bf a, v16bf b, v8f c) {
    return __builtin_amdgcn_wmma_f32_16x16x32_bf16(false, a, false, b,
                                                   (short)0, c, false, false);
}

// ---------- TDM: async 2D tile (128 rows x 32 bf16) global -> LDS ----------
// D# per CDNA5 ISA. pad: every 16 dwords (=64B row)
// insert 4 dwords (16B) -> LDS row stride = 40 shorts (80B, 16B aligned).
DEV void tdm_load_tile_128x32(unsigned lds_byte_addr,
                              const unsigned short* gtile,
                              unsigned tensor_d0, unsigned tensor_d1,
                              unsigned stride0_elems) {
    v4ui g0;
    {
        unsigned long long q0 = ((unsigned long long)lds_byte_addr << 32) | 1ull; // count=1
        unsigned long long q1 =
            ((unsigned long long)(uintptr_t)gtile & 0x01FFFFFFFFFFFFFFull) |
            (2ull << 62);                                                         // type=2
        g0[0] = (unsigned)q0; g0[1] = (unsigned)(q0 >> 32);
        g0[2] = (unsigned)q1; g0[3] = (unsigned)(q1 >> 32);
    }
    v8i g1;
    {
        const unsigned flags = (1u << 16)   // data_size: 2 bytes
                             | (1u << 20)   // pad_enable
                             | (3u << 22)   // pad_interval: 16 dwords
                             | (3u << 25);  // pad_amount: 4 dwords
        unsigned long long q0 = ((unsigned long long)(tensor_d0 & 0xFFFFu) << 48) |
                                (unsigned long long)flags;
        unsigned long long q1 = (32ull << 48)                                   // tile_dim0
                              | ((unsigned long long)tensor_d1 << 16)           // tensor_dim1
                              | ((unsigned long long)tensor_d0 >> 16);
        unsigned long long q2 = ((unsigned long long)stride0_elems << 32)       // dim0 stride
                              | 128ull;                                         // tile_dim1
        unsigned long long q3 = 0ull;
        g1[0] = (int)q0; g1[1] = (int)(q0 >> 32);
        g1[2] = (int)q1; g1[3] = (int)(q1 >> 32);
        g1[4] = (int)q2; g1[5] = (int)(q2 >> 32);
        g1[6] = (int)q3; g1[7] = (int)(q3 >> 32);
    }
    v4i z4; z4[0] = 0; z4[1] = 0; z4[2] = 0; z4[3] = 0;
    v8i z8;
#pragma unroll
    for (int i = 0; i < 8; ++i) z8[i] = 0;
    __builtin_amdgcn_tensor_load_to_lds(g0, g1, z4, z4, z8, 0);
}

// ---------- problem sizes ----------
static constexpr int NROW      = 16384;
static constexpr int LM_DIM    = 4096;
static constexpr int VIS_DIM   = 1024;
static constexpr int CB_DIM    = 256;
static constexpr int NUM_CODES = 8192;
static constexpr int HID       = 512;

// d_out element offsets (return-tuple order)
static constexpr size_t O_LM_Z      = 0;
static constexpr size_t O_VIS_Z     = O_LM_Z      + (size_t)NROW * CB_DIM;
static constexpr size_t O_LM_Q      = O_VIS_Z     + (size_t)NROW * CB_DIM;
static constexpr size_t O_VIS_Q     = O_LM_Q      + (size_t)NROW * CB_DIM;
static constexpr size_t O_LM_IDX    = O_VIS_Q     + (size_t)NROW * CB_DIM;
static constexpr size_t O_VIS_IDX   = O_LM_IDX    + (size_t)NROW;
static constexpr size_t O_LM_COMMIT = O_VIS_IDX   + (size_t)NROW;
static constexpr size_t O_VIS_COMMIT= O_LM_COMMIT + 1;
static constexpr size_t O_LM_RECON  = O_VIS_COMMIT + 1;
static constexpr size_t O_VIS_RECON = O_LM_RECON  + (size_t)NROW * LM_DIM;

// =====================================================================
// bf16 WMMA GEMM with TDM double-buffered staging:
//   C[M,N](f32) = A[M,K](bf16) * Bt[N,K](bf16, N-major) + bias[N]
// Block tile 128x128, 256 threads (8 wave32 waves), wave tile 32x64.
// Requires M%128==0, N%128==0, K%32==0 (all calls satisfy this).
// =====================================================================
__global__ __launch_bounds__(256)
void wmma_gemm_bias(const unsigned short* __restrict__ A,
                    const unsigned short* __restrict__ Bt,
                    const float* __restrict__ bias, float* __restrict__ C,
                    int M, int N, int K) {
    __shared__ __align__(16) unsigned short sA[2][128 * 40];
    __shared__ __align__(16) unsigned short sB[2][128 * 40];

    const int t    = threadIdx.x;
    const int lane = t & 31;
    const int w    = t >> 5;
    const int wm   = w >> 1;          // 0..3 -> 32-row stripe
    const int wn   = w & 1;           // 0..1 -> 64-col stripe
    const int half = lane >> 4;
    const int ln   = lane & 15;
    const int m0   = blockIdx.y * 128;
    const int n0   = blockIdx.x * 128;

    v8f acc[2][4];
#pragma unroll
    for (int mi = 0; mi < 2; ++mi)
#pragma unroll
        for (int ni = 0; ni < 4; ++ni)
#pragma unroll
            for (int i = 0; i < 8; ++i) acc[mi][ni][i] = 0.0f;

    // wave 0 drives the TDM pipeline
    if (w == 0) {
        tdm_load_tile_128x32((unsigned)(uintptr_t)&sA[0][0],
                             A + (size_t)m0 * K, (unsigned)K, (unsigned)M, (unsigned)K);
        tdm_load_tile_128x32((unsigned)(uintptr_t)&sB[0][0],
                             Bt + (size_t)n0 * K, (unsigned)K, (unsigned)N, (unsigned)K);
    }

    int p = 0;
    for (int k0 = 0; k0 < K; k0 += 32) {
        const bool more = (k0 + 32) < K;
        if (w == 0) {
            if (more) {
                tdm_load_tile_128x32((unsigned)(uintptr_t)&sA[p ^ 1][0],
                                     A + (size_t)m0 * K + (k0 + 32),
                                     (unsigned)K, (unsigned)M, (unsigned)K);
                tdm_load_tile_128x32((unsigned)(uintptr_t)&sB[p ^ 1][0],
                                     Bt + (size_t)n0 * K + (k0 + 32),
                                     (unsigned)K, (unsigned)N, (unsigned)K);
                __builtin_amdgcn_s_wait_tensorcnt(2);   // current buffer done
            } else {
                __builtin_amdgcn_s_wait_tensorcnt(0);
            }
        }
        __syncthreads();

        const unsigned short* cA = sA[p];
        const unsigned short* cB = sB[p];
        v16bf af[2], bf[4];
#pragma unroll
        for (int mi = 0; mi < 2; ++mi)
            af[mi] = load_frag_a(cA + (wm * 32 + mi * 16 + ln) * 40 + half * 16);
#pragma unroll
        for (int ni = 0; ni < 4; ++ni)
            bf[ni] = load_frag_b(cB + (wn * 64 + ni * 16 + ln) * 40 + half * 16);
#pragma unroll
        for (int mi = 0; mi < 2; ++mi)
#pragma unroll
            for (int ni = 0; ni < 4; ++ni)
                acc[mi][ni] = wmma_bf16(af[mi], bf[ni], acc[mi][ni]);
        __syncthreads();
        p ^= 1;
    }

    // epilogue: bias + store
#pragma unroll
    for (int mi = 0; mi < 2; ++mi)
#pragma unroll
        for (int ni = 0; ni < 4; ++ni) {
            const int cc = n0 + wn * 64 + ni * 16 + ln;
            const float bs = bias[cc];
#pragma unroll
            for (int i = 0; i < 8; ++i) {
                const int rr = m0 + wm * 32 + mi * 16 + i + 8 * half;
                C[(size_t)rr * N + cc] = acc[mi][ni][i] + bs;
            }
        }
}

// =====================================================================
// Row LayerNorm (+optional ReLU); writes fp32 and optional bf16 copy.
// =====================================================================
__global__ __launch_bounds__(256)
void row_layernorm(const float* __restrict__ in, const float* __restrict__ g,
                   const float* __restrict__ be, float* __restrict__ out,
                   unsigned short* __restrict__ outb, int W, int doRelu) {
    __shared__ float r1[256];
    __shared__ float r2[256];
    const int t = threadIdx.x;
    const size_t row = blockIdx.x;
    const float* src = in + row * W;
    float s = 0.f, ss = 0.f;
    for (int c = t; c < W; c += 256) { float v = src[c]; s += v; ss += v * v; }
    r1[t] = s; r2[t] = ss;
    __syncthreads();
    for (int sr = 128; sr > 0; sr >>= 1) {
        if (t < sr) { r1[t] += r1[t + sr]; r2[t] += r2[t + sr]; }
        __syncthreads();
    }
    const float mu  = r1[0] / (float)W;
    const float var = r2[0] / (float)W - mu * mu;
    const float sc  = rsqrtf(var + 1e-5f);
    float* dst = out + row * W;
    unsigned short* dstb = outb ? outb + row * W : nullptr;
    for (int c = t; c < W; c += 256) {
        float v = (src[c] - mu) * sc * g[c] + be[c];
        if (doRelu) v = fmaxf(v, 0.f);
        dst[c] = v;
        if (dstb) dstb[c] = f2bf(v);
    }
}

// =====================================================================
// prep kernels
// =====================================================================
__global__ __launch_bounds__(256)
void convert_bf16(const float* __restrict__ in, unsigned short* __restrict__ out,
                  long long n) {
    for (long long i = (long long)blockIdx.x * 256 + threadIdx.x; i < n;
         i += (long long)gridDim.x * 256)
        out[i] = f2bf(in[i]);
}

// W[K,N] fp32 -> Wt[N,K] bf16
__global__ __launch_bounds__(256)
void transpose_bf16(const float* __restrict__ in, unsigned short* __restrict__ out,
                    int K, int N) {
    const long long total = (long long)K * N;
    for (long long i = (long long)blockIdx.x * 256 + threadIdx.x; i < total;
         i += (long long)gridDim.x * 256) {
        const int k = (int)(i / N), n = (int)(i % N);
        out[(size_t)n * K + k] = f2bf(in[i]);
    }
}

__global__ __launch_bounds__(256)
void cb_norms(const float* __restrict__ cb, float* __restrict__ cbn) {
    const int k = blockIdx.x * 256 + threadIdx.x;
    if (k < NUM_CODES) {
        const float* e = cb + (size_t)k * CB_DIM;
        float s = 0.f;
        for (int j = 0; j < CB_DIM; ++j) { float v = e[j]; s += v * v; }
        cbn[k] = s;
    }
}

__global__ void zero_commits(float* out) {
    out[O_LM_COMMIT] = 0.f;
    out[O_VIS_COMMIT] = 0.f;
}

// =====================================================================
// Fused VQ: block = 16 z-rows vs all 8192 codes.
// A-fragments (16x256 bf16) stay register-resident across the sweep;
// codebook bf16 streamed from L2. Packed-key argmin via ds_min_u64.
// =====================================================================
__global__ __launch_bounds__(256)
void vq_argmin(const float* __restrict__ z,            // [NROW,256] fp32
               const unsigned short* __restrict__ zb,  // [NROW,256] bf16
               const unsigned short* __restrict__ cbb, // [8192,256] bf16
               const float* __restrict__ cbn,          // [8192]
               const float* __restrict__ cb,           // [8192,256] fp32
               int* __restrict__ idxOut, float* __restrict__ qOut,
               unsigned short* __restrict__ qbOut,
               float* __restrict__ commitOut) {
    __shared__ float znorm[16];
    __shared__ unsigned long long bestKey[16];
    __shared__ int bestIdx[16];
    __shared__ float red[256];

    const int t    = threadIdx.x;
    const int lane = t & 31;
    const int w    = t >> 5;
    const int half = lane >> 4;
    const int ln   = lane & 15;
    const int r0   = blockIdx.x * 16;

    if (t < 16) { znorm[t] = 0.f; bestKey[t] = ~0ULL; }
    __syncthreads();

    // ||z||^2 per row (fp32)
    {
        const int row = t >> 4, cb0 = (t & 15) * 16;
        const float* src = z + (size_t)(r0 + row) * CB_DIM + cb0;
        float fv[16];
#pragma unroll
        for (int j = 0; j < 4; ++j) ((float4*)fv)[j] = ((const float4*)src)[j];
        float ss = 0.f;
#pragma unroll
        for (int j = 0; j < 16; ++j) ss += fv[j] * fv[j];
        atomicAdd(&znorm[row], ss);
    }
    __syncthreads();

    // register-resident A fragments: z rows r0..r0+15, all 256 K
    v16bf afr[8];
#pragma unroll
    for (int s = 0; s < 8; ++s)
        afr[s] = load_frag_a(zb + (size_t)(r0 + ln) * CB_DIM + s * 32 + half * 16);

    float zn[8];
#pragma unroll
    for (int i = 0; i < 8; ++i) zn[i] = znorm[i + 8 * half];

    unsigned long long best[8];
#pragma unroll
    for (int i = 0; i < 8; ++i) best[i] = ~0ULL;

    // 8 waves x 16 codes = 128 codes / iter, 64 iters
    for (int it = 0; it < 64; ++it) {
        const int c0 = it * 128 + w * 16;
        v8f acc;
#pragma unroll
        for (int i = 0; i < 8; ++i) acc[i] = 0.0f;
#pragma unroll
        for (int s = 0; s < 8; ++s) {
            v16bf b = load_frag_b(cbb + (size_t)(c0 + ln) * CB_DIM + s * 32 + half * 16);
            acc = wmma_bf16(afr[s], b, acc);
        }
        const int col = c0 + ln;
        const float cn = cbn[col];
#pragma unroll
        for (int i = 0; i < 8; ++i) {
            const float d = zn[i] - 2.0f * acc[i] + cn;
            const unsigned long long key =
                ((unsigned long long)fbits_ordered(d) << 32) | (unsigned int)col;
            best[i] = key < best[i] ? key : best[i];
        }
    }
#pragma unroll
    for (int i = 0; i < 8; ++i)
        atomicMin(&bestKey[i + 8 * half], best[i]);
    __syncthreads();

    if (t < 16) {
        const int idx = (int)(unsigned int)bestKey[t];
        bestIdx[t] = idx;
        idxOut[r0 + t] = idx;
    }
    __syncthreads();

    // gather q (fp32 + bf16), accumulate commit partial
    float part = 0.f;
    {
        const int row = t >> 4, cb0 = (t & 15) * 16;
        const int idx = bestIdx[row];
        const float* e  = cb + (size_t)idx * CB_DIM + cb0;
        const float* zz = z + (size_t)(r0 + row) * CB_DIM + cb0;
        float* qq       = qOut + (size_t)(r0 + row) * CB_DIM + cb0;
        unsigned short* qb = qbOut + (size_t)(r0 + row) * CB_DIM + cb0;
#pragma unroll
        for (int j = 0; j < 16; ++j) {
            const float qv = e[j];
            qq[j] = qv;
            qb[j] = f2bf(qv);
            const float df = zz[j] - qv;
            part += df * df;
        }
    }
    red[t] = part;
    __syncthreads();
    for (int sr = 128; sr > 0; sr >>= 1) {
        if (t < sr) red[t] += red[t + sr];
        __syncthreads();
    }
    if (t == 0)
        atomicAdd(commitOut, red[0] * (1.0f / ((float)NROW * (float)CB_DIM)));
}

// =====================================================================
// host-side launch
// =====================================================================
extern "C" void kernel_launch(void* const* d_in, const int* in_sizes, int n_in,
                              void* d_out, int out_size, void* d_ws, size_t ws_size,
                              hipStream_t stream) {
    const float* lm_x     = (const float*)d_in[0];
    const float* vis_x    = (const float*)d_in[1];
    const float* lm_w1    = (const float*)d_in[2];
    const float* lm_b1    = (const float*)d_in[3];
    const float* lm_g1    = (const float*)d_in[4];
    const float* lm_be1   = (const float*)d_in[5];
    const float* lm_w2    = (const float*)d_in[6];
    const float* lm_b2    = (const float*)d_in[7];
    const float* lm_g2    = (const float*)d_in[8];
    const float* lm_be2   = (const float*)d_in[9];
    const float* vis_w1   = (const float*)d_in[10];
    const float* vis_b1   = (const float*)d_in[11];
    const float* vis_g1   = (const float*)d_in[12];
    const float* vis_be1  = (const float*)d_in[13];
    const float* vis_w2   = (const float*)d_in[14];
    const float* vis_b2   = (const float*)d_in[15];
    const float* vis_g2   = (const float*)d_in[16];
    const float* vis_be2  = (const float*)d_in[17];
    const float* codebook = (const float*)d_in[18];
    const float* dec_lm_w = (const float*)d_in[19];
    const float* dec_lm_b = (const float*)d_in[20];
    const float* dec_vis_w= (const float*)d_in[21];
    const float* dec_vis_b= (const float*)d_in[22];

    float* out = (float*)d_out;

    // ---- workspace carve-up (bf16 operands + fp32 intermediates) ----
    char* ws = (char*)d_ws;
    auto takeUS = [&](size_t n) { unsigned short* p = (unsigned short*)ws; ws += n * 2; return p; };
    auto takeF  = [&](size_t n) { float* p = (float*)ws; ws += n * 4; return p; };

    unsigned short* lm_xb   = takeUS((size_t)NROW * LM_DIM);
    unsigned short* vis_xb  = takeUS((size_t)NROW * VIS_DIM);
    unsigned short* h_b     = takeUS((size_t)NROW * HID);
    unsigned short* z_b     = takeUS((size_t)NROW * CB_DIM);
    unsigned short* q_b     = takeUS((size_t)NROW * CB_DIM);
    unsigned short* lm_w1t  = takeUS((size_t)LM_DIM * HID);
    unsigned short* vis_w1t = takeUS((size_t)VIS_DIM * HID);
    unsigned short* lm_w2t  = takeUS((size_t)HID * CB_DIM);
    unsigned short* vis_w2t = takeUS((size_t)HID * CB_DIM);
    unsigned short* dlm_wt  = takeUS((size_t)CB_DIM * LM_DIM);
    unsigned short* dvis_wt = takeUS((size_t)CB_DIM * VIS_DIM);
    unsigned short* cb_b    = takeUS((size_t)NUM_CODES * CB_DIM);
    float*          h_pre   = takeF((size_t)NROW * HID);
    float*          z_pre   = takeF((size_t)NROW * CB_DIM);
    float*          cbn     = takeF((size_t)NUM_CODES);

    const dim3 blk(256);

    // ---- prep: bf16 conversions, weight transposes, codebook norms ----
    zero_commits<<<1, 1, 0, stream>>>(out);
    convert_bf16<<<4096, blk, 0, stream>>>(lm_x, lm_xb, (long long)NROW * LM_DIM);
    convert_bf16<<<2048, blk, 0, stream>>>(vis_x, vis_xb, (long long)NROW * VIS_DIM);
    convert_bf16<<<1024, blk, 0, stream>>>(codebook, cb_b, (long long)NUM_CODES * CB_DIM);
    transpose_bf16<<<1024, blk, 0, stream>>>(lm_w1, lm_w1t, LM_DIM, HID);
    transpose_bf16<<<512,  blk, 0, stream>>>(vis_w1, vis_w1t, VIS_DIM, HID);
    transpose_bf16<<<256,  blk, 0, stream>>>(lm_w2, lm_w2t, HID, CB_DIM);
    transpose_bf16<<<256,  blk, 0, stream>>>(vis_w2, vis_w2t, HID, CB_DIM);
    transpose_bf16<<<512,  blk, 0, stream>>>(dec_lm_w, dlm_wt, CB_DIM, LM_DIM);
    transpose_bf16<<<256,  blk, 0, stream>>>(dec_vis_w, dvis_wt, CB_DIM, VIS_DIM);
    cb_norms<<<NUM_CODES / 256, blk, 0, stream>>>(codebook, cbn);

    // ---- LM chain: encode -> VQ -> decode ----
    wmma_gemm_bias<<<dim3(HID / 128, NROW / 128), blk, 0, stream>>>(
        lm_xb, lm_w1t, lm_b1, h_pre, NROW, HID, LM_DIM);
    row_layernorm<<<NROW, blk, 0, stream>>>(h_pre, lm_g1, lm_be1, h_pre, h_b, HID, 1);
    wmma_gemm_bias<<<dim3(CB_DIM / 128, NROW / 128), blk, 0, stream>>>(
        h_b, lm_w2t, lm_b2, z_pre, NROW, CB_DIM, HID);
    row_layernorm<<<NROW, blk, 0, stream>>>(z_pre, lm_g2, lm_be2,
                                            out + O_LM_Z, z_b, CB_DIM, 0);
    vq_argmin<<<NROW / 16, blk, 0, stream>>>(
        out + O_LM_Z, z_b, cb_b, cbn, codebook,
        (int*)(out + O_LM_IDX), out + O_LM_Q, q_b, out + O_LM_COMMIT);
    wmma_gemm_bias<<<dim3(LM_DIM / 128, NROW / 128), blk, 0, stream>>>(
        q_b, dlm_wt, dec_lm_b, out + O_LM_RECON, NROW, LM_DIM, CB_DIM);

    // ---- VIS chain (reuses h_pre/z_pre/h_b/z_b/q_b) ----
    wmma_gemm_bias<<<dim3(HID / 128, NROW / 128), blk, 0, stream>>>(
        vis_xb, vis_w1t, vis_b1, h_pre, NROW, HID, VIS_DIM);
    row_layernorm<<<NROW, blk, 0, stream>>>(h_pre, vis_g1, vis_be1, h_pre, h_b, HID, 1);
    wmma_gemm_bias<<<dim3(CB_DIM / 128, NROW / 128), blk, 0, stream>>>(
        h_b, vis_w2t, vis_b2, z_pre, NROW, CB_DIM, HID);
    row_layernorm<<<NROW, blk, 0, stream>>>(z_pre, vis_g2, vis_be2,
                                            out + O_VIS_Z, z_b, CB_DIM, 0);
    vq_argmin<<<NROW / 16, blk, 0, stream>>>(
        out + O_VIS_Z, z_b, cb_b, cbn, codebook,
        (int*)(out + O_VIS_IDX), out + O_VIS_Q, q_b, out + O_VIS_COMMIT);
    wmma_gemm_bias<<<dim3(VIS_DIM / 128, NROW / 128), blk, 0, stream>>>(
        q_b, dvis_wt, dec_vis_b, out + O_VIS_RECON, NROW, VIS_DIM, CB_DIM);
}
